// Attention_26749056320138
// MI455X (gfx1250) — matrix operand order repaired and checked by reading, not compiled, not run on hardware
//
#include <hip/hip_runtime.h>
#include <hip/hip_bf16.h>

// ---------------------------------------------------------------------------
// MHA forward for gfx1250: bf16 WMMA GEMMs (async-LDS double-buffered) +
// fused flash attention.
// D_MODEL=1024, N_HEADS=16, D_HEAD=64, BATCH=8, SEQ=1024
// ---------------------------------------------------------------------------

typedef __attribute__((ext_vector_type(16))) __bf16 v16bf;
typedef __attribute__((ext_vector_type(8)))  __bf16 v8bf;
typedef __attribute__((ext_vector_type(8)))  float  v8f;
typedef __attribute__((ext_vector_type(4)))  int    v4i;

#define D_MODEL 1024
#define N_HEADS 16
#define D_HEAD  64
#define BATCH   8
#define SEQ     1024
#define MROWS   (BATCH * SEQ)   // 8192

#if defined(__HIP_DEVICE_COMPILE__) && __has_builtin(__builtin_amdgcn_global_load_async_to_lds_b128)
#define HAVE_ASYNC_LDS 1
#else
#define HAVE_ASYNC_LDS 0
#endif

// 16B global -> LDS copy. Async DMA (ASYNCcnt) when available.
static __device__ __forceinline__ void g2l_b128(void* lds, const void* g) {
#if HAVE_ASYNC_LDS
    __builtin_amdgcn_global_load_async_to_lds_b128(
        (__attribute__((address_space(1))) v4i*)g,
        (__attribute__((address_space(3))) v4i*)lds, 0, 0);
#else
    *(v8bf*)lds = *(const v8bf*)g;
#endif
}

static __device__ __forceinline__ void wait_async() {
#if HAVE_ASYNC_LDS
#if __has_builtin(__builtin_amdgcn_s_wait_asynccnt)
    __builtin_amdgcn_s_wait_asynccnt(0);
#else
    asm volatile("s_wait_asynccnt 0x0" ::: "memory");
#endif
#endif
}

static __device__ __forceinline__ v16bf make_a16(const __bf16* c0, const __bf16* c1) {
    v8bf lo = *(const v8bf*)c0;
    v8bf hi = *(const v8bf*)c1;
    v16bf r;
#pragma unroll
    for (int i = 0; i < 8; ++i) { r[i] = lo[i]; r[i + 8] = hi[i]; }
    return r;
}

// ---------------------------------------------------------------------------
// Stage 1: conversions / layout transforms (fp32 -> bf16)
// ---------------------------------------------------------------------------
__global__ void k_cvt_x(const float* __restrict__ src, __bf16* __restrict__ dst) {
    int i = (blockIdx.x * blockDim.x + threadIdx.x) * 4;
    float4 f = *(const float4*)(src + i);
    dst[i + 0] = (__bf16)f.x;
    dst[i + 1] = (__bf16)f.y;
    dst[i + 2] = (__bf16)f.z;
    dst[i + 3] = (__bf16)f.w;
}

// W (H, D, E) -> Wt[(h*64+e)][d]   (B-tile layout: n-major, k-contiguous)
__global__ void k_cvt_wqkv(const float* __restrict__ src, __bf16* __restrict__ dst) {
    int i = blockIdx.x * blockDim.x + threadIdx.x;       // over 1024*1024
    int h = i >> 16;
    int rem = i & 65535;
    int d = rem >> 6;
    int e = rem & 63;
    dst[(size_t)((h << 6) + e) * 1024 + d] = (__bf16)src[i];
}

// W_O (H, E, D) -> Wt[d][(h*64+e)]
__global__ void k_cvt_wo(const float* __restrict__ src, __bf16* __restrict__ dst) {
    int i = blockIdx.x * blockDim.x + threadIdx.x;       // over 1024*1024
    int k = i >> 10;
    int d = i & 1023;
    dst[(size_t)d * 1024 + k] = (__bf16)src[i];
}

// ---------------------------------------------------------------------------
// Stage 2: bf16 GEMM  C(8192 x 1024) = A(8192x1024) * Bt^T + bias
// Block: 256 threads (8 waves). Tile: M=128 (16 rows/wave), N=128.
// Double-buffered LDS with async global->LDS staging.
// ---------------------------------------------------------------------------

// stage one 128x32 A strip and 128x32 B strip into LDS buffer (2 chunks each)
static __device__ __forceinline__ void gemm_stage(
        __bf16 (*As)[40], __bf16 (*Bs)[40],
        const __bf16* __restrict__ A, const __bf16* __restrict__ Bt,
        int mbase, int nbase, int k0, int t) {
    {   // A chunk 0 and 1
        int row = t >> 2, kq = (t & 3) * 8;
        g2l_b128(&As[row][kq], A + (size_t)(mbase + row) * 1024 + k0 + kq);
        g2l_b128(&As[row + 64][kq], A + (size_t)(mbase + row + 64) * 1024 + k0 + kq);
    }
    {   // B chunk 0 and 1
        int n = t >> 2, kq = (t & 3) * 8;
        g2l_b128(&Bs[n][kq], Bt + (size_t)(nbase + n) * 1024 + k0 + kq);
        g2l_b128(&Bs[n + 64][kq], Bt + (size_t)(nbase + n + 64) * 1024 + k0 + kq);
    }
}

// one 32-deep k-step: depth-2 pipelined B fragments, 8 WMMAs
static __device__ __forceinline__ void gemm_step(
        const __bf16 (*As)[40], const __bf16 (*Bs)[40],
        int rowl, int koffA, int koffB, int lane15, v8f acc[8]) {
    v16bf a = make_a16(&As[rowl][koffA], &As[rowl][koffA + 8]);
    v16bf bcur = make_a16(&Bs[lane15][koffB], &Bs[lane15][koffB + 8]);
#pragma unroll
    for (int tn = 0; tn < 8; ++tn) {
        v16bf bnext;
        if (tn < 7) {
            int nl = (tn + 1) * 16 + lane15;
            bnext = make_a16(&Bs[nl][koffB], &Bs[nl][koffB + 8]);
        }
        acc[tn] = __builtin_amdgcn_wmma_f32_16x16x32_bf16(
            false, a, false, bcur, (short)0, acc[tn], false, false);
        if (tn < 7) bcur = bnext;
    }
}

__global__ __launch_bounds__(256, 2)
void k_gemm_qkv(const __bf16* __restrict__ A, const __bf16* __restrict__ Bt,
                const float* __restrict__ bias, __bf16* __restrict__ outQKV) {
    __shared__ __bf16 As[2][128][40];
    __shared__ __bf16 Bs[2][128][40];

    const int nbase = blockIdx.x * 128;
    const int mbase = blockIdx.y * 128;
    const int t = threadIdx.x;
    const int wave = t >> 5, lane = t & 31;
    const int lhalf = (lane < 16);
    const int lane15 = lane & 15;
    const int rowl = wave * 16 + lane15;
    const int koffA = lhalf ? 0 : 16;
    const int koffB = lhalf ? 0 : 16;

    v8f acc[8] = {};

    gemm_stage(As[0], Bs[0], A, Bt, mbase, nbase, 0, t);
    wait_async();
    __syncthreads();

    int buf = 0;
    for (int k0 = 32; k0 < 1024; k0 += 32) {
        gemm_stage(As[buf ^ 1], Bs[buf ^ 1], A, Bt, mbase, nbase, k0, t);
        gemm_step(As[buf], Bs[buf], rowl, koffA, koffB, lane15, acc);
        wait_async();
        __syncthreads();
        buf ^= 1;
    }
    gemm_step(As[buf], Bs[buf], rowl, koffA, koffB, lane15, acc);

#pragma unroll
    for (int tn = 0; tn < 8; ++tn) {
#pragma unroll
        for (int j = 0; j < 8; ++j) {
            int r = wave * 16 + (lhalf ? j : j + 8);
            int c = tn * 16 + lane15;
            int m = mbase + r;
            int n = nbase + c;
            float val = acc[tn][j] + bias[n];
            int bb = m >> 10, s = m & 1023, h = n >> 6, e = n & 63;
            outQKV[((size_t)(bb * 16 + h) * 1024 + s) * 64 + e] = (__bf16)val;
        }
    }
}

__global__ __launch_bounds__(256, 2)
void k_gemm_out(const __bf16* __restrict__ A, const __bf16* __restrict__ Bt,
                const float* __restrict__ bias, float* __restrict__ out) {
    __shared__ __bf16 As[2][128][40];
    __shared__ __bf16 Bs[2][128][40];

    const int nbase = blockIdx.x * 128;
    const int mbase = blockIdx.y * 128;
    const int t = threadIdx.x;
    const int wave = t >> 5, lane = t & 31;
    const int lhalf = (lane < 16);
    const int lane15 = lane & 15;
    const int rowl = wave * 16 + lane15;
    const int koffA = lhalf ? 0 : 16;
    const int koffB = lhalf ? 0 : 16;

    v8f acc[8] = {};

    gemm_stage(As[0], Bs[0], A, Bt, mbase, nbase, 0, t);
    wait_async();
    __syncthreads();

    int buf = 0;
    for (int k0 = 32; k0 < 1024; k0 += 32) {
        gemm_stage(As[buf ^ 1], Bs[buf ^ 1], A, Bt, mbase, nbase, k0, t);
        gemm_step(As[buf], Bs[buf], rowl, koffA, koffB, lane15, acc);
        wait_async();
        __syncthreads();
        buf ^= 1;
    }
    gemm_step(As[buf], Bs[buf], rowl, koffA, koffB, lane15, acc);

#pragma unroll
    for (int tn = 0; tn < 8; ++tn) {
#pragma unroll
        for (int j = 0; j < 8; ++j) {
            int r = wave * 16 + (lhalf ? j : j + 8);
            int c = tn * 16 + lane15;
            int m = mbase + r;
            int n = nbase + c;
            out[(size_t)m * 1024 + n] = acc[tn][j] + bias[n];
        }
    }
}

// ---------------------------------------------------------------------------
// Stage 3: fused flash attention per (b, h, 64-query tile)
//   S = Q K^T / 8, online softmax, Z = P V.  Key tiles of 64.
//   8 waves: wave w -> rows 16*(w&3), e/key cols 32*(w>>2)+{0,16}.
// ---------------------------------------------------------------------------
__global__ __launch_bounds__(256, 2)
void k_attn(const __bf16* __restrict__ Q, const __bf16* __restrict__ K,
            const __bf16* __restrict__ V, const int* __restrict__ mask,
            __bf16* __restrict__ Z) {
    __shared__ __bf16 Qs[64][72];
    __shared__ __bf16 Ks[64][72];
    __shared__ __bf16 Vts[64][72];   // [e][key]
    __shared__ __bf16 Ps[64][72];    // [q][key]
    __shared__ float  Ss[64][68];
    __shared__ float  alphaL[64];
    __shared__ float  invL[64];
    __shared__ int    mL[64];

    const int qtile = blockIdx.x;    // 16 tiles of 64 rows
    const int h = blockIdx.y;
    const int bb = blockIdx.z;
    const int qbase = qtile * 64;

    const int t = threadIdx.x;
    const int wave = t >> 5, lane = t & 31;
    const int lhalf = (lane < 16);
    const int lane15 = lane & 15;
    const int rbase = 16 * (wave & 3);
    const int nbase = 32 * (wave >> 2);
    const int rowl = rbase + lane15;
    const int koffA = lhalf ? 0 : 16;
    const int koffB = lhalf ? 0 : 16;

    const size_t bh = (size_t)(bb * 16 + h) * SEQ * 64;
    const __bf16* Qbh = Q + bh;
    const __bf16* Kbh = K + bh;
    const __bf16* Vbh = V + bh;

    // load Q tile once (async -> LDS), 2 chunks/thread
    {
        int row = t >> 3, eq = (t & 7) * 8;
        g2l_b128(&Qs[row][eq], Qbh + (size_t)(qbase + row) * 64 + eq);
        g2l_b128(&Qs[row + 32][eq], Qbh + (size_t)(qbase + row + 32) * 64 + eq);
    }

    float mreg = -__builtin_inff();
    float lreg = 0.0f;
    v8f zacc[2] = {};

    for (int kt = 0; kt < SEQ / 64; ++kt) {
        const int kbase = kt * 64;
        // --- stage K (async), V^T (explicit transpose), mask into LDS ---
        {
            int row = t >> 3, eq = (t & 7) * 8;
            g2l_b128(&Ks[row][eq], Kbh + (size_t)(kbase + row) * 64 + eq);
            g2l_b128(&Ks[row + 32][eq], Kbh + (size_t)(kbase + row + 32) * 64 + eq);
        }
        {
            int key = t >> 3, eq = (t & 7) * 8;
            v8bf v0 = *(const v8bf*)(Vbh + (size_t)(kbase + key) * 64 + eq);
            v8bf v1 = *(const v8bf*)(Vbh + (size_t)(kbase + key + 32) * 64 + eq);
#pragma unroll
            for (int i = 0; i < 8; ++i) Vts[eq + i][key] = v0[i];
#pragma unroll
            for (int i = 0; i < 8; ++i) Vts[eq + i][key + 32] = v1[i];
        }
        if (t < 64) mL[t] = mask[bb * SEQ + kbase + t];
        wait_async();
        __syncthreads();

        // --- S tile: Q (16x64) x K^T (64x16 per n-tile) ---
        v8f sacc[2] = {};
#pragma unroll
        for (int k0 = 0; k0 < 64; k0 += 32) {
            v16bf a = make_a16(&Qs[rowl][k0 + koffA], &Qs[rowl][k0 + koffA + 8]);
            v16bf b0 = make_a16(&Ks[nbase + lane15][k0 + koffB],
                                &Ks[nbase + lane15][k0 + koffB + 8]);
            v16bf b1 = make_a16(&Ks[nbase + 16 + lane15][k0 + koffB],
                                &Ks[nbase + 16 + lane15][k0 + koffB + 8]);
            sacc[0] = __builtin_amdgcn_wmma_f32_16x16x32_bf16(
                false, a, false, b0, (short)0, sacc[0], false, false);
            sacc[1] = __builtin_amdgcn_wmma_f32_16x16x32_bf16(
                false, a, false, b1, (short)0, sacc[1], false, false);
        }
#pragma unroll
        for (int tn = 0; tn < 2; ++tn) {
#pragma unroll
            for (int j = 0; j < 8; ++j) {
                int r = rbase + (lhalf ? j : j + 8);
                int c = nbase + tn * 16 + lane15;
                Ss[r][c] = sacc[tn][j];
            }
        }
        __syncthreads();

        // --- online softmax: one thread per query row ---
        if (t < 64) {
            float mx = mreg;
#pragma unroll 8
            for (int j = 0; j < 64; ++j) {
                float s = Ss[t][j] * 0.125f;
                if (mL[j] == 0) s = -__builtin_inff();
                Ss[t][j] = s;
                mx = fmaxf(mx, s);
            }
            float al = __expf(mreg - mx);
            float ln = lreg * al;
#pragma unroll 8
            for (int j = 0; j < 64; ++j) {
                float p = __expf(Ss[t][j] - mx);
                ln += p;
                Ps[t][j] = (__bf16)p;
            }
            mreg = mx;
            lreg = ln;
            alphaL[t] = al;
        }
        __syncthreads();

        // --- rescale accumulators, then Z += P V ---
#pragma unroll
        for (int tn = 0; tn < 2; ++tn) {
#pragma unroll
            for (int j = 0; j < 8; ++j) {
                int r = rbase + (lhalf ? j : j + 8);
                zacc[tn][j] *= alphaL[r];
            }
        }
#pragma unroll
        for (int k0 = 0; k0 < 64; k0 += 32) {
            v16bf a = make_a16(&Ps[rowl][k0 + koffA], &Ps[rowl][k0 + koffA + 8]);
            v16bf b0 = make_a16(&Vts[nbase + lane15][k0 + koffB],
                                &Vts[nbase + lane15][k0 + koffB + 8]);
            v16bf b1 = make_a16(&Vts[nbase + 16 + lane15][k0 + koffB],
                                &Vts[nbase + 16 + lane15][k0 + koffB + 8]);
            zacc[0] = __builtin_amdgcn_wmma_f32_16x16x32_bf16(
                false, a, false, b0, (short)0, zacc[0], false, false);
            zacc[1] = __builtin_amdgcn_wmma_f32_16x16x32_bf16(
                false, a, false, b1, (short)0, zacc[1], false, false);
        }
        __syncthreads();
    }

    if (t < 64) invL[t] = (lreg > 0.0f) ? (1.0f / lreg) : 0.0f;
    __syncthreads();

    // --- write Z in (B, S, H*E) bf16 layout for the output GEMM ---
#pragma unroll
    for (int tn = 0; tn < 2; ++tn) {
#pragma unroll
        for (int j = 0; j < 8; ++j) {
            int r = rbase + (lhalf ? j : j + 8);
            int e = nbase + tn * 16 + lane15;
            float val = zacc[tn][j] * invL[r];
            int s = qbase + r;
            Z[(size_t)(bb * SEQ + s) * D_MODEL + h * 64 + e] = (__bf16)val;
        }
    }
}

// ---------------------------------------------------------------------------
// Host-side launch
// ---------------------------------------------------------------------------
extern "C" void kernel_launch(void* const* d_in, const int* in_sizes, int n_in,
                              void* d_out, int out_size, void* d_ws, size_t ws_size,
                              hipStream_t stream) {
    const float* x  = (const float*)d_in[0];
    const float* WQ = (const float*)d_in[1];
    const float* WK = (const float*)d_in[2];
    const float* WV = (const float*)d_in[3];
    const float* WO = (const float*)d_in[4];
    const float* bQ = (const float*)d_in[5];
    const float* bK = (const float*)d_in[6];
    const float* bV = (const float*)d_in[7];
    const float* bO = (const float*)d_in[8];
    const int* maskp = (const int*)d_in[9];
    float* out = (float*)d_out;

    char* ws = (char*)d_ws;
    __bf16* xb   = (__bf16*)ws;  ws += (size_t)MROWS * D_MODEL * 2;
    __bf16* wq_t = (__bf16*)ws;  ws += (size_t)D_MODEL * D_MODEL * 2;
    __bf16* wk_t = (__bf16*)ws;  ws += (size_t)D_MODEL * D_MODEL * 2;
    __bf16* wv_t = (__bf16*)ws;  ws += (size_t)D_MODEL * D_MODEL * 2;
    __bf16* wo_t = (__bf16*)ws;  ws += (size_t)D_MODEL * D_MODEL * 2;
    __bf16* qb   = (__bf16*)ws;  ws += (size_t)MROWS * D_MODEL * 2;
    __bf16* kb   = (__bf16*)ws;  ws += (size_t)MROWS * D_MODEL * 2;
    __bf16* vb   = (__bf16*)ws;  ws += (size_t)MROWS * D_MODEL * 2;
    __bf16* zb   = (__bf16*)ws;  ws += (size_t)MROWS * D_MODEL * 2;

    // 1) conversions / layout transforms
    k_cvt_x<<<(MROWS * D_MODEL) / (256 * 4), 256, 0, stream>>>(x, xb);
    k_cvt_wqkv<<<(D_MODEL * D_MODEL) / 256, 256, 0, stream>>>(WQ, wq_t);
    k_cvt_wqkv<<<(D_MODEL * D_MODEL) / 256, 256, 0, stream>>>(WK, wk_t);
    k_cvt_wqkv<<<(D_MODEL * D_MODEL) / 256, 256, 0, stream>>>(WV, wv_t);
    k_cvt_wo  <<<(D_MODEL * D_MODEL) / 256, 256, 0, stream>>>(WO, wo_t);

    // 2) QKV projections (M=128, N=128 tiles)
    dim3 ggrid(D_MODEL / 128, MROWS / 128);
    k_gemm_qkv<<<ggrid, 256, 0, stream>>>(xb, wq_t, bQ, qb);
    k_gemm_qkv<<<ggrid, 256, 0, stream>>>(xb, wk_t, bK, kb);
    k_gemm_qkv<<<ggrid, 256, 0, stream>>>(xb, wv_t, bV, vb);

    // 3) fused flash attention
    dim3 agrid(SEQ / 64, N_HEADS, BATCH);
    k_attn<<<agrid, 256, 0, stream>>>(qb, kb, vb, maskp, zb);

    // 4) output projection (fp32 out + bias)
    k_gemm_out<<<ggrid, 256, 0, stream>>>(zb, wo_t, bO, out);
}
